// Attention_7069516169290
// MI455X (gfx1250) — matrix operand order repaired and checked by reading, not compiled, not run on hardware
//
#include <hip/hip_runtime.h>

// ---------------------------------------------------------------------------
// Attention forward for MI455X (gfx1250, wave32, WMMA).
//   x      [2, 2048, 1024]  fp32
//   w_qkv  [1024, 3072]     fp32
//   out    [2, 2048, 1024]  fp32
// Kernel 1: QKV = x @ w_qkv  (bf16 WMMA, fp32 accum) -> workspace bf16.
//           Q pre-scaled by d^-0.5; Q,K stored [b,h,n,d]; V stored TRANSPOSED
//           [b,h,d,n] so the flash kernel can stage it with straight
//           async global->LDS row copies (no in-LDS transpose needed).
// Kernel 2: flash attention per (b,h): online-softmax(Q K^T) V with WMMA;
//           K/V tiles staged via GLOBAL_LOAD_ASYNC_TO_LDS_B128.
// ---------------------------------------------------------------------------

typedef __attribute__((ext_vector_type(16))) __bf16 v16bf;
typedef __attribute__((ext_vector_type(8)))  float  v8f;
typedef __attribute__((ext_vector_type(4)))  int    v4i;

#define HEADS 16
#define DH    64
#define BATCH 2
#define SEQ   2048
#define DIMK  1024
#define QKV_COLS (3 * HEADS * DH)                       // 3072
#define HELEMS ((size_t)BATCH * HEADS * SEQ * DH)       // 4,194,304 per tensor
#define KSTR  72                                        // LDS row stride (shorts): 144B, 16B-aligned

// --- CDNA5 async global->LDS path (guarded; falls back to manual copy) -----
#if defined(__has_builtin)
#  if __has_builtin(__builtin_amdgcn_global_load_async_to_lds_b128)
#    define HAVE_ASYNC_LDS 1
#  endif
#  if __has_builtin(__builtin_amdgcn_s_wait_asynccnt)
#    define HAVE_WAIT_ASYNC 1
#  endif
#endif

#if defined(HAVE_ASYNC_LDS)
typedef __attribute__((address_space(3))) v4i las_v4i;   // LDS-qualified int4

__device__ __forceinline__ void async_copy_b128(const void* gsrc, void* ldst) {
  // Signature (from hipcc diagnostic): param0 is generic `int __vector(4)*`.
  // LDS arg passed AS3-qualified: exact match if param1 is AS3, implicit
  // AS3->generic conversion if param1 is generic.
  __builtin_amdgcn_global_load_async_to_lds_b128(
      (v4i*)const_cast<void*>(gsrc), (las_v4i*)ldst, 0, 0);
}
#endif

__device__ __forceinline__ void async_wait_all() {
#if defined(HAVE_ASYNC_LDS) && defined(HAVE_WAIT_ASYNC)
  __builtin_amdgcn_s_wait_asynccnt(0);
#endif
}

__device__ __forceinline__ unsigned short f2bfu(float f) {
  unsigned u = __float_as_uint(f);
  u += 0x7FFFu + ((u >> 16) & 1u);          // round-to-nearest-even
  return (unsigned short)(u >> 16);
}
__device__ __forceinline__ __bf16 u2bf(unsigned short s) {
  union { unsigned short u; __bf16 b; } v; v.u = s; return v.b;
}

// ---------------------------------------------------------------------------
// Kernel 1: QKV projection GEMM.  Grid (3072/64, 4096/128), block 256 (8 waves).
// Each wave owns a 16x64 strip: 4 WMMA accumulators, K stepped by 32.
// ---------------------------------------------------------------------------
__global__ __launch_bounds__(256) void qkv_gemm_kernel(
    const float* __restrict__ x, const float* __restrict__ w,
    unsigned short* __restrict__ qkv_ws) {
  __shared__ unsigned short As[128][34];   // [m][k]  bf16 bits, padded
  __shared__ unsigned short Bs[64][34];    // [n][k]  bf16 bits (W transposed)

  const int t    = threadIdx.x;
  const int wave = t >> 5;
  const int lane = t & 31;
  const int lrow = lane & 15;
  const int lhi  = lane >> 4;
  const int m0   = blockIdx.y * 128;
  const int c0   = blockIdx.x * 64;

  v8f acc[4];
#pragma unroll
  for (int nt = 0; nt < 4; nt++)
#pragma unroll
    for (int r = 0; r < 8; r++) acc[nt][r] = 0.0f;

  const int arow = t >> 1;               // 0..127
  const int ac0  = (t & 1) * 16;         // 0 or 16
  const int bk   = t >> 3;               // 0..31
  const int bn0  = (t & 7) * 8;          // 0..56

  for (int k0 = 0; k0 < DIMK; k0 += 32) {
    __syncthreads();
    // stage A tile 128x32 (fp32 -> bf16)
    const float* xp = x + (size_t)(m0 + arow) * DIMK + k0 + ac0;
#pragma unroll
    for (int i = 0; i < 16; i++) As[arow][ac0 + i] = f2bfu(xp[i]);
    // stage B tile 32x64, transposed into [n][k]
    const float* wp = w + (size_t)(k0 + bk) * QKV_COLS + c0 + bn0;
#pragma unroll
    for (int i = 0; i < 8; i++) Bs[bn0 + i][bk] = f2bfu(wp[i]);
    __syncthreads();

    // A fragment: 16x32, row = lane&15 (+wave strip), K split per half-wave
    v16bf a;
    const int row = wave * 16 + lrow;
    const int kh  = lhi * 16;
#pragma unroll
    for (int i = 0; i < 8; i++) {
      a[i]     = u2bf(As[row][kh + i]);
      a[8 + i] = u2bf(As[row][kh + 8 + i]);
    }
#pragma unroll
    for (int nt = 0; nt < 4; nt++) {
      v16bf bfr;
      const int n  = nt * 16 + lrow;
      const int kb = lhi * 16;
#pragma unroll
      for (int i = 0; i < 16; i++) bfr[i] = u2bf(Bs[n][kb + i]);
      acc[nt] = __builtin_amdgcn_wmma_f32_16x16x32_bf16(
          false, a, false, bfr, (short)0, acc[nt], false, false);
    }
  }

  // store: map column c -> (which q/k/v, head h, d); row m -> (b, n)
  const float scale = 0.125f;   // DH^-0.5 folded into Q
#pragma unroll
  for (int nt = 0; nt < 4; nt++) {
    const int c0t   = c0 + nt * 16;
    const int which = c0t >> 10;        // 0=q 1=k 2=v (16-col tile never straddles)
    const int cc    = c0t & 1023;
    const int h     = cc >> 6;
    const int d     = (cc & 63) + lrow;
    const float sc  = (which == 0) ? scale : 1.0f;
    unsigned short* dst = qkv_ws + (size_t)which * HELEMS;
#pragma unroll
    for (int r = 0; r < 8; r++) {
      const int m = m0 + wave * 16 + r + 8 * lhi;
      const int b = m >> 11;
      const int n = m & 2047;
      const size_t bhbase = ((size_t)b * HEADS + h);
      // V is stored transposed [b,h,d,n]; Q,K stored [b,h,n,d]
      const size_t idx = (which == 2)
          ? (bhbase * DH + d) * SEQ + n
          : (bhbase * SEQ + n) * DH + d;
      dst[idx] = f2bfu(acc[nt][r] * sc);
    }
  }
}

// ---------------------------------------------------------------------------
// Kernel 2: flash attention.  Grid (2048/64 q-blocks, 32 bh), block 128 (4 waves).
// Each wave owns 16 q rows; loop over 64-key blocks with online softmax.
// K block rows ([key][d]) and V^T block rows ([d][key]) are contiguous 128B
// runs -> staged with async b128 copies (4 chunks per thread, 512 total/tile).
// ---------------------------------------------------------------------------
__global__ __launch_bounds__(128) void flash_attn_kernel(
    const unsigned short* __restrict__ qws, const unsigned short* __restrict__ kws,
    const unsigned short* __restrict__ vtws, float* __restrict__ out) {
  __shared__ unsigned short Ks[64][KSTR];       // [key][d]
  __shared__ unsigned short Vt[64][KSTR];       // [d][key]
  __shared__ unsigned short Ps[4][16][KSTR];    // per-wave P tile [m][key]

  const int t    = threadIdx.x;
  const int wave = t >> 5;
  const int lane = t & 31;
  const int lrow = lane & 15;
  const int lhi  = lane >> 4;
  const int bh   = blockIdx.y;
  const int b    = bh >> 4;
  const int h    = bh & 15;
  const int q0   = blockIdx.x * 64;

  const unsigned short* Q  = qws  + (size_t)bh * SEQ * DH;
  const unsigned short* K  = kws  + (size_t)bh * SEQ * DH;
  const unsigned short* VT = vtws + (size_t)bh * DH * SEQ;   // [d][n]

  // Q fragments for this wave's 16 rows (two k-steps over d=64), kept in VGPRs
  v16bf qa[2];
  {
    const int row = q0 + wave * 16 + lrow;
    const int kh  = lhi * 16;
#pragma unroll
    for (int st = 0; st < 2; st++) {
      const unsigned short* qp = Q + (size_t)row * DH + st * 32;
#pragma unroll
      for (int i = 0; i < 8; i++) {
        qa[st][i]     = u2bf(qp[kh + i]);
        qa[st][8 + i] = u2bf(qp[kh + 8 + i]);
      }
    }
  }

  float mrun[8], lrun[8];
#pragma unroll
  for (int r = 0; r < 8; r++) { mrun[r] = -1e30f; lrun[r] = 0.0f; }
  v8f o[4];
#pragma unroll
  for (int dt = 0; dt < 4; dt++)
#pragma unroll
    for (int r = 0; r < 8; r++) o[dt][r] = 0.0f;

  for (int j0 = 0; j0 < SEQ; j0 += 64) {
    __syncthreads();
    // stage K block [64 keys][64 d] and V^T block [64 d][64 keys]:
    // 512 x 16B chunks each; thread t handles chunks t*4 .. t*4+3.
#pragma unroll
    for (int c = 0; c < 4; c++) {
      const int l    = t * 4 + c;       // 0..511
      const int row  = l >> 3;          // 0..63
      const int cb   = (l & 7) * 16;    // byte offset within the 128B row
      const char* ksrc = (const char*)(K  + (size_t)(j0 + row) * DH) + cb;
      const char* vsrc = (const char*)(VT + (size_t)row * SEQ + j0)  + cb;
      char* kdst = (char*)(&Ks[0][0]) + row * (KSTR * 2) + cb;
      char* vdst = (char*)(&Vt[0][0]) + row * (KSTR * 2) + cb;
#if defined(HAVE_ASYNC_LDS)
      async_copy_b128(ksrc, kdst);
      async_copy_b128(vsrc, vdst);
#else
      *(uint4*)kdst = *(const uint4*)ksrc;
      *(uint4*)vdst = *(const uint4*)vsrc;
#endif
    }
    if (j0 + 64 < SEQ) {  // hint next K/V^T blocks toward the caches
      __builtin_prefetch(K + (size_t)(j0 + 64 + (t >> 1)) * DH, 0, 1);
      __builtin_prefetch(VT + (size_t)(t >> 1) * SEQ + j0 + 64, 0, 1);
    }
    async_wait_all();
    __syncthreads();

    // S = Q K^T  (16 x 64 per wave): 4 key tiles x 2 k-steps
    v8f sacc[4];
#pragma unroll
    for (int nt = 0; nt < 4; nt++) {
#pragma unroll
      for (int r = 0; r < 8; r++) sacc[nt][r] = 0.0f;
      const int n = nt * 16 + lrow;          // key index in block
#pragma unroll
      for (int st = 0; st < 2; st++) {
        v16bf bfr;
        const int kb = st * 32 + lhi * 16;
#pragma unroll
        for (int i = 0; i < 16; i++) bfr[i] = u2bf(Ks[n][kb + i]);
        sacc[nt] = __builtin_amdgcn_wmma_f32_16x16x32_bf16(
            false, qa[st], false, bfr, (short)0, sacc[nt], false, false);
      }
    }

    // online softmax (row reductions over the 16-lane half-wave)
    float alpha[8];
#pragma unroll
    for (int r = 0; r < 8; r++) {
      float mx = fmaxf(fmaxf(sacc[0][r], sacc[1][r]), fmaxf(sacc[2][r], sacc[3][r]));
#pragma unroll
      for (int off = 1; off < 16; off <<= 1) mx = fmaxf(mx, __shfl_xor(mx, off, 32));
      const float mnew = fmaxf(mrun[r], mx);
      alpha[r] = __expf(mrun[r] - mnew);
      mrun[r]  = mnew;
      float rs = 0.0f;
#pragma unroll
      for (int nt = 0; nt < 4; nt++) {
        const float p = __expf(sacc[nt][r] - mnew);
        sacc[nt][r] = p;
        rs += p;
      }
#pragma unroll
      for (int off = 1; off < 16; off <<= 1) rs += __shfl_xor(rs, off, 32);
      lrun[r] = lrun[r] * alpha[r] + rs;
#pragma unroll
      for (int dt = 0; dt < 4; dt++) o[dt][r] *= alpha[r];
    }

    // P (C-layout) -> LDS as bf16 so it can be re-read in A-fragment layout
#pragma unroll
    for (int nt = 0; nt < 4; nt++)
#pragma unroll
      for (int r = 0; r < 8; r++)
        Ps[wave][r + 8 * lhi][nt * 16 + lrow] = f2bfu(sacc[nt][r]);
    __syncthreads();

    // O += P V  : 4 d tiles x 2 k-steps over the 64 keys
#pragma unroll
    for (int dt = 0; dt < 4; dt++) {
      const int n = dt * 16 + lrow;          // d index
#pragma unroll
      for (int st = 0; st < 2; st++) {
        v16bf a, bfr;
        const int kh = st * 32 + lhi * 16;
#pragma unroll
        for (int i = 0; i < 8; i++) {
          a[i]     = u2bf(Ps[wave][lrow][kh + i]);
          a[8 + i] = u2bf(Ps[wave][lrow][kh + 8 + i]);
        }
        const int kb = st * 32 + lhi * 16;
#pragma unroll
        for (int i = 0; i < 16; i++) bfr[i] = u2bf(Vt[n][kb + i]);
        o[dt] = __builtin_amdgcn_wmma_f32_16x16x32_bf16(
            false, a, false, bfr, (short)0, o[dt], false, false);
      }
    }
  }

  // epilogue: normalize and store fp32 [b, n, h*d]
#pragma unroll
  for (int r = 0; r < 8; r++) {
    const float inv = 1.0f / lrun[r];
    const int m = q0 + wave * 16 + r + 8 * lhi;
#pragma unroll
    for (int dt = 0; dt < 4; dt++) {
      const int d = dt * 16 + lrow;
      out[(((size_t)b * SEQ + m) * (HEADS * DH)) + h * DH + d] = o[dt][r] * inv;
    }
  }
}

// ---------------------------------------------------------------------------
extern "C" void kernel_launch(void* const* d_in, const int* in_sizes, int n_in,
                              void* d_out, int out_size, void* d_ws, size_t ws_size,
                              hipStream_t stream) {
  const float* x = (const float*)d_in[0];
  const float* w = (const float*)d_in[1];
  float* out = (float*)d_out;

  unsigned short* ws = (unsigned short*)d_ws;   // bf16 bit patterns
  unsigned short* qv = ws;                      // Q [b,h,n,d]
  unsigned short* kv = ws + HELEMS;             // K [b,h,n,d]
  unsigned short* vt = ws + 2 * HELEMS;         // V^T [b,h,d,n]

  qkv_gemm_kernel<<<dim3(QKV_COLS / 64, (BATCH * SEQ) / 128), 256, 0, stream>>>(x, w, ws);
  flash_attn_kernel<<<dim3(SEQ / 64, BATCH * HEADS), 128, 0, stream>>>(qv, kv, vt, out);
}
